// LRU_1305670058048
// MI455X (gfx1250) — matrix operand order repaired and checked by reading, not compiled, not run on hardware
//
#include <hip/hip_runtime.h>
#include <hip/hip_bf16.h>

typedef __bf16 bf16t;
typedef __attribute__((ext_vector_type(16))) __bf16 v16bf;
typedef __attribute__((ext_vector_type(8)))  __bf16 v8bf;
typedef __attribute__((ext_vector_type(8)))  float  v8f;

#define B_SZ   16
#define T_SZ   2048
#define DIN    512
#define DOUT   512
#define NST    512
#define M_SZ   (B_SZ * T_SZ)      /* 32768 */
#define CHUNK  128
#define NCHUNK (T_SZ / CHUNK)     /* 16 */

// ---------------------------------------------------------------------------
// CDNA5 async global->LDS copy of 16 bytes per lane (ASYNCcnt tracked).
// The VDST VGPR holds the wave-relative LDS byte address; per the ISA
// aperture rules the low 32 bits of a generic pointer to LDS are exactly
// that offset.
// ---------------------------------------------------------------------------
__device__ __forceinline__ void async_copy16(const bf16t* g, bf16t* l) {
  unsigned laddr = (unsigned)(uintptr_t)l;
  asm volatile("global_load_async_to_lds_b128 %0, %1, off"
               :: "v"(laddr), "v"(g) : "memory");
}
__device__ __forceinline__ void wait_async0() {
  asm volatile("s_wait_asynccnt 0x0" ::: "memory");
}

// ---------------------------------------------------------------------------
// lam = exp(-exp(nu_log)) * (cos(exp(theta_log)) + i sin(exp(theta_log)))
// lam buffer layout: [0..511]=re, [512..1023]=im
// ---------------------------------------------------------------------------
__global__ void lru_prep_params(const float* __restrict__ nu_log,
                                const float* __restrict__ theta_log,
                                float* __restrict__ lam) {
  int n = blockIdx.x * blockDim.x + threadIdx.x;
  if (n >= NST) return;
  float mag = expf(-expf(nu_log[n]));
  float ph  = expf(theta_log[n]);
  lam[n]        = mag * cosf(ph);
  lam[NST + n]  = mag * sinf(ph);
}

// ---------------------------------------------------------------------------
// Weight conversion to bf16:
//   Wcat [2N, DIN]  : rows 0..511 = gamma*B_re, rows 512..1023 = gamma*B_im
//   WC   [3*DOUT*NST]: [0..]=C_re, [262144..]= -C_im, [524288..]= D
// ---------------------------------------------------------------------------
__global__ void lru_convert_weights(const float* __restrict__ B_re,
                                    const float* __restrict__ B_im,
                                    const float* __restrict__ C_re,
                                    const float* __restrict__ C_im,
                                    const float* __restrict__ Dm,
                                    const float* __restrict__ gamma_log,
                                    bf16t* __restrict__ Wcat,
                                    bf16t* __restrict__ WC) {
  int id = blockIdx.x * blockDim.x + threadIdx.x;     // 5 * 262144 threads
  int region = id >> 18;
  int idx    = id & 262143;
  int n      = idx >> 9;
  if (region == 0) {
    Wcat[idx]          = (bf16t)(B_re[idx] * expf(gamma_log[n]));
  } else if (region == 1) {
    Wcat[262144 + idx] = (bf16t)(B_im[idx] * expf(gamma_log[n]));
  } else if (region == 2) {
    WC[idx]            = (bf16t)(C_re[idx]);
  } else if (region == 3) {
    WC[262144 + idx]   = (bf16t)(-C_im[idx]);
  } else {
    WC[524288 + idx]   = (bf16t)(Dm[idx]);
  }
}

// input f32 [M, DIN] -> bf16, 4 elements per thread
__global__ void lru_convert_input(const float* __restrict__ in,
                                  bf16t* __restrict__ out) {
  size_t id = (size_t)(blockIdx.x * blockDim.x + threadIdx.x) * 4;
  float4 v = *(const float4*)(in + id);
  out[id + 0] = (bf16t)v.x;
  out[id + 1] = (bf16t)v.y;
  out[id + 2] = (bf16t)v.z;
  out[id + 3] = (bf16t)v.w;
}

// ---------------------------------------------------------------------------
// bf16 GEMM, out[M,N] = sum_p A_p[M,K] * W_p[N,K]^T   (row-major, f32 out)
// block tile 128x128, BK=64, 8 waves -> each wave 32x64 (2x4 WMMA tiles),
// 16 WMMAs per stage. Global->LDS staging via async-to-LDS (ASYNCcnt),
// double buffered: one s_wait_asynccnt + one barrier per 64-deep K slab.
// ---------------------------------------------------------------------------
__global__ __launch_bounds__(256) void lru_gemm_bf16_nt(
    const bf16t* __restrict__ A0, const bf16t* __restrict__ W0,
    const bf16t* __restrict__ A1, const bf16t* __restrict__ W1,
    const bf16t* __restrict__ A2, const bf16t* __restrict__ W2,
    int nPairs, float* __restrict__ out, int M, int N, int K) {
  __shared__ __align__(32) bf16t As[2][128 * 64];
  __shared__ __align__(32) bf16t Ws[2][128 * 64];

  const int tid    = threadIdx.x;
  const int lane   = tid & 31;
  const int waveId = tid >> 5;
  const int wm     = waveId >> 1;     // 0..3  (M direction)
  const int wn     = waveId & 1;      // 0..1  (N direction)
  const int m16    = lane & 15;
  const int hi     = lane >> 4;

  const int blockN = blockIdx.x * 128;
  const int blockM = blockIdx.y * 128;

  const int lrow = tid >> 1;          // 0..127
  const int lcol = (tid & 1) * 32;    // 0 or 32 (each thread: 2x16 elements)

  const int ksteps = K >> 6;          // BK = 64
  const int total  = nPairs * ksteps;

  v8f acc[2][4];
  for (int mi = 0; mi < 2; ++mi)
    for (int ni = 0; ni < 4; ++ni)
      for (int e = 0; e < 8; ++e) acc[mi][ni][e] = 0.0f;

  // preload stage 0 (pair 0, kk = 0)
  {
    size_t aoff = (size_t)(blockM + lrow) * K + lcol;
    size_t woff = (size_t)(blockN + lrow) * K + lcol;
    async_copy16(A0 + aoff,      &As[0][lrow * 64 + lcol]);
    async_copy16(A0 + aoff + 16, &As[0][lrow * 64 + lcol + 16]);
    async_copy16(W0 + woff,      &Ws[0][lrow * 64 + lcol]);
    async_copy16(W0 + woff + 16, &Ws[0][lrow * 64 + lcol + 16]);
  }
  wait_async0();
  __syncthreads();

  int buf = 0;
  for (int s = 0; s < total; ++s) {
    const bool havenext = (s + 1) < total;
    if (havenext) {
      int sn = s + 1;
      int p  = sn / ksteps;
      int kk = (sn - p * ksteps) << 6;
      const bf16t* Ap = (p == 0) ? A0 : ((p == 1) ? A1 : A2);
      const bf16t* Wp = (p == 0) ? W0 : ((p == 1) ? W1 : W2);
      size_t aoff = (size_t)(blockM + lrow) * K + kk + lcol;
      size_t woff = (size_t)(blockN + lrow) * K + kk + lcol;
      async_copy16(Ap + aoff,      &As[buf ^ 1][lrow * 64 + lcol]);
      async_copy16(Ap + aoff + 16, &As[buf ^ 1][lrow * 64 + lcol + 16]);
      async_copy16(Wp + woff,      &Ws[buf ^ 1][lrow * 64 + lcol]);
      async_copy16(Wp + woff + 16, &Ws[buf ^ 1][lrow * 64 + lcol + 16]);
    }

    for (int ks = 0; ks < 64; ks += 32) {
      // A fragments: lane halves hold K 0-15 / 16-31, contiguous
      v16bf af[2];
      for (int mi = 0; mi < 2; ++mi) {
        int row = wm * 32 + mi * 16 + m16;
        union { v16bf v; v8bf h[2]; } u;
        u.h[0] = *(const v8bf*)(&As[buf][row * 64 + ks + hi * 16]);
        u.h[1] = *(const v8bf*)(&As[buf][row * 64 + ks + hi * 16 + 8]);
        af[mi] = u.v;
      }
      // B fragments: lane halves hold K 0-15 / 16-31, contiguous
      v16bf bfr[4];
      for (int ni = 0; ni < 4; ++ni) {
        int row = wn * 64 + ni * 16 + m16;
        bfr[ni] = *(const v16bf*)(&Ws[buf][row * 64 + ks + hi * 16]);
      }
      for (int mi = 0; mi < 2; ++mi)
        for (int ni = 0; ni < 4; ++ni)
          acc[mi][ni] = __builtin_amdgcn_wmma_f32_16x16x32_bf16(
              false, af[mi], false, bfr[ni], (short)0, acc[mi][ni], false, false);
    }

    if (havenext) {
      wait_async0();        // this thread's next-stage LDS writes landed
      __syncthreads();      // everyone's writes visible, reads of buf done
      buf ^= 1;
    }
  }

  // C/D layout: VGPR g, lanes 0-15 -> M=g, lanes 16-31 -> M=8+g; N = lane&15
  for (int mi = 0; mi < 2; ++mi)
    for (int ni = 0; ni < 4; ++ni) {
      int col     = blockN + wn * 64 + ni * 16 + m16;
      int rowbase = blockM + wm * 32 + mi * 16 + hi * 8;
      for (int g = 0; g < 8; ++g)
        out[(size_t)(rowbase + g) * N + col] = acc[mi][ni][g];
    }
}

// ---------------------------------------------------------------------------
// Scan pass 1: per (b, chunk, n) local scan of CHUNK steps (no carry).
// Bu [M, 2N] (cols 0..511 = re, 512..1023 = im) overwritten in place with the
// chunk-local states; chunk end state stored in carry[b*16*512 + chunk*512+n].
// ---------------------------------------------------------------------------
__global__ void lru_scan_pass1(float* __restrict__ Bu,
                               const float* __restrict__ lam,
                               float2* __restrict__ carry) {
  int id    = blockIdx.x * blockDim.x + threadIdx.x;   // 131072
  int n     = id & 511;
  int chunk = (id >> 9) & 15;
  int b     = id >> 13;
  float lre = lam[n], lim = lam[NST + n];
  float sre = 0.f, sim = 0.f;
  size_t r = ((size_t)b * T_SZ + (size_t)chunk * CHUNK) * (2 * NST);
  for (int t = 0; t < CHUNK; ++t) {
    float ure = Bu[r + n];
    float uim = Bu[r + NST + n];
    float nre = lre * sre - lim * sim + ure;
    float nim = lre * sim + lim * sre + uim;
    sre = nre; sim = nim;
    Bu[r + n]       = sre;
    Bu[r + NST + n] = sim;
    r += 2 * NST;
  }
  carry[id] = make_float2(sre, sim);
}

// Scan pass 2: per (b, n) exclusive scan over the 16 chunk summaries.
__global__ void lru_scan_pass2(const float* __restrict__ lam,
                               float2* __restrict__ carry) {
  int id = blockIdx.x * blockDim.x + threadIdx.x;   // 8192
  int n  = id & 511;
  int b  = id >> 9;
  float lre = lam[n], lim = lam[NST + n];
  float ar = lre, ai = lim;                 // lam^128 via 7 squarings
  for (int i = 0; i < 7; ++i) {
    float nr = ar * ar - ai * ai;
    float ni = 2.f * ar * ai;
    ar = nr; ai = ni;
  }
  float cre = 0.f, cim = 0.f;
  for (int c = 0; c < NCHUNK; ++c) {
    int idx = (b * NCHUNK + c) * NST + n;
    float2 e = carry[idx];
    carry[idx] = make_float2(cre, cim);     // exclusive carry-in
    float nr = ar * cre - ai * cim + e.x;
    float ni = ar * cim + ai * cre + e.y;
    cre = nr; cim = ni;
  }
}

// Scan pass 3: states[t] = local[t] + lam^(t_off+1) * carry_in; emit bf16.
__global__ void lru_scan_pass3(const float* __restrict__ Bu,
                               const float* __restrict__ lam,
                               const float2* __restrict__ carry,
                               bf16t* __restrict__ sre_o,
                               bf16t* __restrict__ sim_o) {
  int id    = blockIdx.x * blockDim.x + threadIdx.x;   // 131072
  int n     = id & 511;
  int chunk = (id >> 9) & 15;
  int b     = id >> 13;
  float lre = lam[n], lim = lam[NST + n];
  float2 c  = carry[id];
  float pre = lre, pim = lim;
  size_t row = (size_t)b * T_SZ + (size_t)chunk * CHUNK;
  size_t r = row * (2 * NST);
  for (int t = 0; t < CHUNK; ++t) {
    float sre = Bu[r + n]       + pre * c.x - pim * c.y;
    float sim = Bu[r + NST + n] + pre * c.y + pim * c.x;
    sre_o[(row + t) * NST + n] = (bf16t)sre;
    sim_o[(row + t) * NST + n] = (bf16t)sim;
    float nr = pre * lre - pim * lim;
    float ni = pre * lim + pim * lre;
    pre = nr; pim = ni;
    r += 2 * NST;
  }
}

// ---------------------------------------------------------------------------
extern "C" void kernel_launch(void* const* d_in, const int* in_sizes, int n_in,
                              void* d_out, int out_size, void* d_ws, size_t ws_size,
                              hipStream_t stream) {
  const float* input     = (const float*)d_in[0];
  const float* Dm        = (const float*)d_in[1];
  const float* nu_log    = (const float*)d_in[2];
  const float* theta_log = (const float*)d_in[3];
  const float* gamma_log = (const float*)d_in[4];
  const float* B_re      = (const float*)d_in[5];
  const float* B_im      = (const float*)d_in[6];
  const float* C_re      = (const float*)d_in[7];
  const float* C_im      = (const float*)d_in[8];
  float* out = (float*)d_out;

  char* ws = (char*)d_ws;
  float*  lam   = (float*)(ws);                       //    4 KB
  bf16t*  Wcat  = (bf16t*)(ws + 4096);                //    1 MB  [2N, DIN]
  bf16t*  WC    = (bf16t*)(ws + 1052672);             //  1.5 MB  C_re | -C_im | D
  bf16t*  inBf  = (bf16t*)(ws + 2625536);             //   32 MB  [M, DIN]
  float*  Bu    = (float*)(ws + 36179968);            //  128 MB  [M, 2N]
  float2* carry = (float2*)(ws + 170397696);          //    1 MB
  bf16t*  sre   = (bf16t*)(ws + 171446272);           //   32 MB  [M, N]
  bf16t*  sim   = (bf16t*)(ws + 205000704);           //   32 MB  [M, N]

  lru_prep_params<<<2, 256, 0, stream>>>(nu_log, theta_log, lam);
  lru_convert_weights<<<(5 * 262144) / 256, 256, 0, stream>>>(
      B_re, B_im, C_re, C_im, Dm, gamma_log, Wcat, WC);
  lru_convert_input<<<(M_SZ * DIN / 4) / 256, 256, 0, stream>>>(input, inBf);

  // Bu[M, 2N] = input_bf16 @ [gamma*B_re ; gamma*B_im]^T
  lru_gemm_bf16_nt<<<dim3((2 * NST) / 128, M_SZ / 128), 256, 0, stream>>>(
      inBf, Wcat, nullptr, nullptr, nullptr, nullptr, 1, Bu, M_SZ, 2 * NST, DIN);

  lru_scan_pass1<<<(B_SZ * NCHUNK * NST) / 256, 256, 0, stream>>>(Bu, lam, carry);
  lru_scan_pass2<<<(B_SZ * NST) / 256, 256, 0, stream>>>(lam, carry);
  lru_scan_pass3<<<(B_SZ * NCHUNK * NST) / 256, 256, 0, stream>>>(Bu, lam, carry, sre, sim);

  // y = s_re @ C_re^T + s_im @ (-C_im)^T + input @ D^T
  lru_gemm_bf16_nt<<<dim3(DOUT / 128, M_SZ / 128), 256, 0, stream>>>(
      sre, WC, sim, WC + 262144, inBf, WC + 524288, 3, out, M_SZ, DOUT, DIN);
}
